// PCD_Align_69973607187172
// MI455X (gfx1250) — matrix operand order repaired and checked by reading, not compiled, not run on hardware
//
#include <hip/hip_runtime.h>
#include <hip/hip_bf16.h>
#include <cstdint>
#include <cstddef>

#define NF    64
#define BATCH 4
#define DGRP  8

typedef __attribute__((ext_vector_type(16))) _Float16 v16h;
typedef __attribute__((ext_vector_type(8)))  _Float16 v8h;
typedef __attribute__((ext_vector_type(8)))  float    v8f;

__device__ __forceinline__ float lrelu01(float v) { return v > 0.f ? v : 0.1f * v; }
__device__ __forceinline__ int   iclamp(int v, int lo, int hi) { return v < lo ? lo : (v > hi ? hi : v); }

// ---------------------------------------------------------------------------
// Pack conv weights [Cout,Cin,3,3] f32 -> WMMA A-fragment layout f16:
//   dst[mt][kc][lane][16]  (16x32 f16 A, contiguous k = 16*h + e in lane half h)
// kOrder==0: K = tap*Cin + c   (3x3 conv, tap-major K)
// kOrder==1: K = c*9 + tap     (DCN einsum ordering over (g,c,k))
// ---------------------------------------------------------------------------
__global__ void pack_wfrag(const float* __restrict__ w, _Float16* __restrict__ dst,
                           int Cout, int Cin, int K, int kcCount, int kOrder)
{
    int idx = blockIdx.x * blockDim.x + threadIdx.x;
    int total = ((Cout + 15) / 16) * kcCount * 512;
    if (idx >= total) return;
    int e    = idx & 15;
    int lane = (idx >> 4) & 31;
    int kc   = (idx >> 9) % kcCount;
    int mt   = idx / (kcCount << 9);
    int r = lane & 15, hh = lane >> 4;
    int v = e >> 1, p = e & 1;
    int Kloc = 16 * hh + 2 * v + p;  // A-matrix 16x32 f16, contiguous k = 16*h + e (matches B loads)
    int Kg = kc * 32 + Kloc;
    int co = mt * 16 + r;
    float val = 0.f;
    if (co < Cout && Kg < K) {
        int c, tap;
        if (kOrder == 0) { tap = Kg / Cin; c = Kg % Cin; }
        else             { c = Kg / 9;     tap = Kg % 9; }
        int ky = tap / 3, kx = tap % 3;
        val = w[(((size_t)co * Cin + c) * 3 + ky) * 3 + kx];
    }
    dst[idx] = (_Float16)val;
}

// ---------------------------------------------------------------------------
// f32 NCHW -> f16 NHWC with 1-pixel zero pad ring (ring pre-zeroed by memset)
// ---------------------------------------------------------------------------
__global__ void nchw_to_nhwc_f16(const float* __restrict__ src, _Float16* __restrict__ dst,
                                 int H, int W, int Csrc, int Ctot, int coff)
{
    int idx = blockIdx.x * blockDim.x + threadIdx.x;
    int total = BATCH * Csrc * H * W;
    if (idx >= total) return;
    int x = idx % W; int t = idx / W;
    int y = t % H;   t /= H;
    int c = t % Csrc;
    int b = t / Csrc;
    dst[(((size_t)b * (H + 2) + (y + 1)) * (W + 2) + (x + 1)) * Ctot + coff + c] = (_Float16)src[idx];
}

// ---------------------------------------------------------------------------
// 2x bilinear upsample (half-pixel centers, edge clamp), NHWC-pad f16 in/out,
// writes 64 channels at channel offset coff, with scalar scale.
// ---------------------------------------------------------------------------
__global__ void upsample2x_f16(const _Float16* __restrict__ src, int hs, int ws,
                               _Float16* __restrict__ dst, int Ctot, int coff, float scale)
{
    int hd = 2 * hs, wd = 2 * ws;
    int idx = blockIdx.x * blockDim.x + threadIdx.x;
    int total = BATCH * hd * wd * (NF / 8);
    if (idx >= total) return;
    int c8 = idx & 7; int t = idx >> 3;
    int x = t % wd; t /= wd;
    int y = t % hd;
    int b = t / hd;

    float sy = (y + 0.5f) * 0.5f - 0.5f;
    float sx = (x + 0.5f) * 0.5f - 0.5f;
    float y0f = floorf(sy), x0f = floorf(sx);
    float fy = sy - y0f,    fx = sx - x0f;
    int y0 = iclamp((int)y0f, 0, hs - 1), y1 = iclamp((int)y0f + 1, 0, hs - 1);
    int x0 = iclamp((int)x0f, 0, ws - 1), x1 = iclamp((int)x0f + 1, 0, ws - 1);

    const _Float16* sb = src + (size_t)c8 * 8;
    auto at = [&](int yy, int xx) -> v8h {
        return *(const v8h*)(sb + (((size_t)b * (hs + 2) + (yy + 1)) * (ws + 2) + (xx + 1)) * NF);
    };
    v8h p00 = at(y0, x0), p01 = at(y0, x1), p10 = at(y1, x0), p11 = at(y1, x1);
    float w00 = (1.f - fy) * (1.f - fx), w01 = (1.f - fy) * fx;
    float w10 = fy * (1.f - fx),         w11 = fy * fx;
    v8h o;
    #pragma unroll
    for (int j = 0; j < 8; ++j) {
        float v = w00 * (float)p00[j] + w01 * (float)p01[j] + w10 * (float)p10[j] + w11 * (float)p11[j];
        o[j] = (_Float16)(v * scale);
    }
    *(v8h*)(dst + (((size_t)b * (hd + 2) + (y + 1)) * (wd + 2) + (x + 1)) * Ctot + coff + c8 * 8) = o;
}

// ---------------------------------------------------------------------------
// DCNv2 gather: x (NHWC-pad f16, channels 0..63 of a Ctot=128 buffer),
// off216 (NHWC-pad f16, C=216: [0:144]=offsets, [144:216]=mask logits).
// Writes val[( b*H*W + pix )*576 + g*72 + c*9 + k] = bilinear(x_gc) * sigmoid(m)
// i.e. GEMM-B ready with K = (g*8+c)*9 + k.
// ---------------------------------------------------------------------------
__global__ void dcn_gather(const _Float16* __restrict__ xin, int Ctot,
                           const _Float16* __restrict__ off216,
                           _Float16* __restrict__ val, int H, int W)
{
    int idx = blockIdx.x * blockDim.x + threadIdx.x;
    int total = BATCH * H * W * DGRP;
    if (idx >= total) return;
    int g = idx % DGRP; int t = idx / DGRP;
    int pix = t % (H * W);
    int b = t / (H * W);
    int y = pix / W, x = pix % W;

    const _Float16* op = off216 + (((size_t)b * (H + 2) + (y + 1)) * (W + 2) + (x + 1)) * 216;
    _Float16* vb = val + ((size_t)b * H * W + pix) * 576 + g * 72;
    const _Float16* xb = xin + (size_t)g * 8;

    v8h z;
    #pragma unroll
    for (int j = 0; j < 8; ++j) z[j] = (_Float16)0.f;

    #pragma unroll
    for (int k = 0; k < 9; ++k) {
        float dy = (float)op[2 * (g * 9 + k)];
        float dx = (float)op[2 * (g * 9 + k) + 1];
        float ml = (float)op[144 + g * 9 + k];
        float m  = 1.f / (1.f + __expf(-ml));
        float py = (float)y + (float)(k / 3 - 1) + dy;
        float px = (float)x + (float)(k % 3 - 1) + dx;
        float y0f = floorf(py), x0f = floorf(px);
        int y0 = (int)y0f, x0 = (int)x0f;
        float wy = py - y0f, wx = px - x0f;
        float w00 = (1.f - wy) * (1.f - wx), w01 = (1.f - wy) * wx;
        float w10 = wy * (1.f - wx),         w11 = wy * wx;
        auto ld = [&](int yy, int xx) -> v8h {
            if (yy < 0 || yy >= H || xx < 0 || xx >= W) return z;
            return *(const v8h*)(xb + (((size_t)b * (H + 2) + (yy + 1)) * (W + 2) + (xx + 1)) * Ctot);
        };
        v8h v00 = ld(y0, x0), v01 = ld(y0, x0 + 1), v10 = ld(y0 + 1, x0), v11 = ld(y0 + 1, x0 + 1);
        #pragma unroll
        for (int c = 0; c < 8; ++c) {
            float v = w00 * (float)v00[c] + w01 * (float)v01[c]
                    + w10 * (float)v10[c] + w11 * (float)v11[c];
            vb[c * 9 + k] = (_Float16)(v * m);
        }
    }
}

// ---------------------------------------------------------------------------
// WMMA implicit-GEMM conv / GEMM kernel, compile-time shapes.
//   CIN:  input channels (conv) or K (flat GEMM)
//   CCPT: Cin/32 chunks per tap (0 => flat GEMM over val[pix][K])
//   KCC:  total K chunks (K/32)
// A staged in LDS once per block (shared by 4 waves); B is one aligned
// 32-byte load per lane per chunk at a constant immediate offset from one of
// three per-row base pointers -> the unrolled loop is ds_read_b128 x2 +
// global_load_b128 x2 + v_wmma per chunk with no per-iteration index math.
// ---------------------------------------------------------------------------
template<int CIN, int CCPT, int KCC>
__global__ __launch_bounds__(128)
void conv16_wmma(const _Float16* __restrict__ in,
                 const _Float16* __restrict__ wfrag,
                 const float* __restrict__ bias,
                 float* __restrict__ out32,
                 _Float16* __restrict__ out16,
                 int H, int W, int Cout,
                 int act, int outCtot, int outCoff)
{
    __shared__ __align__(32) _Float16 aLds[KCC * 512];

    int Mt = (Cout + 15) / 16;
    int mt = blockIdx.z % Mt;
    int b  = blockIdx.z / Mt;

    // stage this cout-tile's A fragments into LDS (all K chunks)
    {
        const uint4* wsrc = (const uint4*)(wfrag + (size_t)mt * KCC * 512);
        uint4* wdst = (uint4*)aLds;
        #pragma unroll 4
        for (int i = threadIdx.x; i < KCC * 64; i += 128) wdst[i] = wsrc[i];
    }
    __syncthreads();

    int wave = threadIdx.x >> 5, lane = threadIdx.x & 31;
    int Wt = W >> 4;
    int xt = blockIdx.x * 4 + wave;
    if (xt >= Wt) return;
    int x0 = xt << 4;
    int y  = blockIdx.y;
    int n  = lane & 15, h = lane >> 4;

    const _Float16* aBase = aLds + (lane << 4);
    v8f acc = {};

    if constexpr (CCPT > 0) {
        // 3x3 conv over NHWC-pad input; three row base pointers, constant offsets
        size_t rowStride = (size_t)(W + 2) * CIN;
        const _Float16* row0 = in + (((size_t)b * (H + 2) + y) * (W + 2) + (x0 + n)) * CIN + (h << 4);
        const _Float16* row1 = row0 + rowStride;
        const _Float16* row2 = row1 + rowStride;
        __builtin_prefetch(row1, 0, 1);   // rows for taps 3..8: warm them early
        __builtin_prefetch(row2, 0, 1);
        const _Float16* rows[3] = { row0, row1, row2 };
        #pragma unroll
        for (int tap = 0; tap < 9; ++tap) {
            const _Float16* rp = rows[tap / 3] + (tap % 3) * CIN;
            #pragma unroll
            for (int cc = 0; cc < CCPT; ++cc) {
                const int kc = tap * CCPT + cc;
                v16h a  = *(const v16h*)(aBase + kc * 512);
                v16h bf = *(const v16h*)(rp + cc * 32);
                acc = __builtin_amdgcn_wmma_f32_16x16x32_f16(false, a, false, bf,
                                                             (short)0, acc, false, false);
            }
        }
    } else {
        // flat GEMM over val[pix][K]
        const _Float16* bp = in + ((size_t)b * H * W + (size_t)y * W + x0 + n) * CIN + (h << 4);
        __builtin_prefetch(bp, 0, 1);
        #pragma unroll
        for (int kc = 0; kc < KCC; ++kc) {
            v16h a  = *(const v16h*)(aBase + kc * 512);
            v16h bf = *(const v16h*)(bp + kc * 32);
            acc = __builtin_amdgcn_wmma_f32_16x16x32_f16(false, a, false, bf,
                                                         (short)0, acc, false, false);
        }
    }

    int coBase = mt << 4;
    float rr[8];
    #pragma unroll
    for (int j = 0; j < 8; ++j) {
        int ch = coBase + j + 8 * h;
        float r = acc[j];
        r = (ch < Cout) ? r + bias[ch] : 0.f;
        if (act) r = lrelu01(r);
        rr[j] = r;
    }

    if (out32) {
        #pragma unroll
        for (int j = 0; j < 8; ++j) {
            int ch = coBase + j + 8 * h;
            if (ch < Cout)
                out32[((size_t)b * outCtot + outCoff + ch) * H * W + (size_t)y * W + x0 + n] = rr[j];
        }
    } else {
        size_t pix = ((size_t)b * (H + 2) + (y + 1)) * (W + 2) + (x0 + n + 1);
        _Float16* op = out16 + pix * (size_t)outCtot + outCoff + coBase + 8 * h;
        if (((outCtot & 7) == 0) && (coBase + 16 <= Cout)) {
            v8h pk;
            #pragma unroll
            for (int j = 0; j < 8; ++j) pk[j] = (_Float16)rr[j];
            *(v8h*)op = pk;
        } else {
            #pragma unroll
            for (int j = 0; j < 8; ++j) {
                int ch = coBase + j + 8 * h;
                if (ch < Cout) op[j] = (_Float16)rr[j];
            }
        }
    }
}

// ---------------------------------------------------------------------------
// Host orchestration
// ---------------------------------------------------------------------------
extern "C" void kernel_launch(void* const* d_in, const int* in_sizes, int n_in,
                              void* d_out, int out_size, void* d_ws, size_t ws_size,
                              hipStream_t stream)
{
    (void)in_sizes; (void)n_in; (void)out_size; (void)ws_size;

    // params flattened as sorted-key pytree after the 6 feature tensors
    auto PW = [&](int k) { return (const float*)d_in[6 + 2 * k]; };
    auto PB = [&](int k) { return (const float*)d_in[7 + 2 * k]; };

    uintptr_t base = (uintptr_t)d_ws;
    size_t off = 0;
    auto alloc = [&](size_t bytes) -> void* {
        off = (off + 255) & ~(size_t)255;
        void* p = (void*)(base + off);
        off += bytes;
        return p;
    };
    auto f16buf = [&](int H, int W, int C) {
        return (_Float16*)alloc((size_t)BATCH * (H + 2) * (W + 2) * C * 2);
    };

    // activation buffers (f16 NHWC, 1-px pad ring; ring zeroed by one memset)
    _Float16 *cat3 = f16buf(32, 32, 128), *t3a = f16buf(32, 32, 64), *off3 = f16buf(32, 32, 64),
             *o216_3 = f16buf(32, 32, 216), *fea3 = f16buf(32, 32, 64);
    _Float16 *cat2 = f16buf(64, 64, 128), *t2cat = f16buf(64, 64, 128), *t2b = f16buf(64, 64, 64),
             *off2 = f16buf(64, 64, 64), *o216_2 = f16buf(64, 64, 216),
             *dcat2 = f16buf(64, 64, 128), *fea2 = f16buf(64, 64, 64);
    _Float16 *cat1 = f16buf(128, 128, 128), *t1cat = f16buf(128, 128, 128), *t1b = f16buf(128, 128, 64),
             *off1 = f16buf(128, 128, 64), *o216_1 = f16buf(128, 128, 216),
             *dcat1 = f16buf(128, 128, 128);
    size_t actEnd = off;
    (void)hipMemsetAsync(d_ws, 0, actEnd, stream);

    _Float16* val = (_Float16*)alloc((size_t)BATCH * 128 * 128 * 576 * 2);

    auto wfragBuf = [&](int Cout, int K) {
        return (_Float16*)alloc((size_t)((Cout + 15) / 16) * (K / 32) * 512 * 2);
    };
    _Float16 *wf_oc1_3 = wfragBuf(64, 1152), *wf_oc2_3 = wfragBuf(64, 576),
             *wf_doff_3 = wfragBuf(216, 576), *wf_dcn_3 = wfragBuf(64, 576);
    _Float16 *wf_oc1_2 = wfragBuf(64, 1152), *wf_oc2_2 = wfragBuf(64, 1152),
             *wf_oc3_2 = wfragBuf(64, 576), *wf_doff_2 = wfragBuf(216, 576),
             *wf_dcn_2 = wfragBuf(64, 576), *wf_fea_2 = wfragBuf(64, 1152);
    _Float16 *wf_oc1_1 = wfragBuf(64, 1152), *wf_oc2_1 = wfragBuf(64, 1152),
             *wf_oc3_1 = wfragBuf(64, 576), *wf_doff_1 = wfragBuf(216, 576),
             *wf_dcn_1 = wfragBuf(64, 576), *wf_fea_1 = wfragBuf(64, 1152);

    auto packW = [&](const float* w, _Float16* dst, int Cout, int Cin, int K, int kOrder) {
        int kcCount = K / 32;
        int total = ((Cout + 15) / 16) * kcCount * 512;
        pack_wfrag<<<(total + 255) / 256, 256, 0, stream>>>(w, dst, Cout, Cin, K, kcCount, kOrder);
    };
    auto toNHWC = [&](const float* src, _Float16* dst, int H, int W, int Csrc, int Ctot, int coff) {
        int total = BATCH * Csrc * H * W;
        nchw_to_nhwc_f16<<<(total + 255) / 256, 256, 0, stream>>>(src, dst, H, W, Csrc, Ctot, coff);
    };
    // mode 0: conv Cin=128 | mode 1: conv Cin=64 | mode 2: flat GEMM K=576
    auto conv = [&](const _Float16* in, const _Float16* wf, const float* bias,
                    float* o32, _Float16* o16, int H, int W, int Cout, int mode,
                    int act, int Ctot, int coff) {
        int Mt = (Cout + 15) / 16;
        dim3 g((W / 16 + 3) / 4, H, BATCH * Mt);
        if (mode == 0)
            conv16_wmma<128, 4, 36><<<g, 128, 0, stream>>>(in, wf, bias, o32, o16, H, W, Cout, act, Ctot, coff);
        else if (mode == 1)
            conv16_wmma<64, 2, 18><<<g, 128, 0, stream>>>(in, wf, bias, o32, o16, H, W, Cout, act, Ctot, coff);
        else
            conv16_wmma<576, 0, 18><<<g, 128, 0, stream>>>(in, wf, bias, o32, o16, H, W, Cout, act, Ctot, coff);
    };
    auto up2 = [&](const _Float16* src, int hs, int ws, _Float16* dst, int Ctot, int coff, float scale) {
        int total = BATCH * (2 * hs) * (2 * ws) * (NF / 8);
        upsample2x_f16<<<(total + 255) / 256, 256, 0, stream>>>(src, hs, ws, dst, Ctot, coff, scale);
    };
    auto gather = [&](const _Float16* x, const _Float16* o216, int H, int W) {
        int total = BATCH * H * W * DGRP;
        dcn_gather<<<(total + 255) / 256, 256, 0, stream>>>(x, 128, o216, val, H, W);
    };

    for (int sx = 0; sx < 2; ++sx) {
        const float* a1 = (const float*)d_in[sx == 0 ? 0 : 3];
        const float* a2 = (const float*)d_in[sx == 0 ? 1 : 4];
        const float* a3 = (const float*)d_in[sx == 0 ? 2 : 5];
        const float* b1 = (const float*)d_in[sx == 0 ? 3 : 0];
        const float* b2 = (const float*)d_in[sx == 0 ? 4 : 1];
        const float* b3 = (const float*)d_in[sx == 0 ? 5 : 2];

        // pack this direction's weights into fragment layout
        packW(PW(28 + sx),     wf_oc1_3, 64, 128, 1152, 0);
        packW(PW(30 + sx),     wf_oc2_3, 64, 64,  576,  0);
        packW(PW(25 + 2 * sx), wf_doff_3, 216, 64, 576, 0);
        packW(PW(24 + 2 * sx), wf_dcn_3, 64, 64,  576,  1);
        packW(PW(18 + sx),     wf_oc1_2, 64, 128, 1152, 0);
        packW(PW(20 + sx),     wf_oc2_2, 64, 128, 1152, 0);
        packW(PW(22 + sx),     wf_oc3_2, 64, 64,  576,  0);
        packW(PW(13 + 2 * sx), wf_doff_2, 216, 64, 576, 0);
        packW(PW(12 + 2 * sx), wf_dcn_2, 64, 64,  576,  1);
        packW(PW(16 + sx),     wf_fea_2, 64, 128, 1152, 0);
        packW(PW(6 + sx),      wf_oc1_1, 64, 128, 1152, 0);
        packW(PW(8 + sx),      wf_oc2_1, 64, 128, 1152, 0);
        packW(PW(10 + sx),     wf_oc3_1, 64, 64,  576,  0);
        packW(PW(1 + 2 * sx),  wf_doff_1, 216, 64, 576, 0);
        packW(PW(0 + 2 * sx),  wf_dcn_1, 64, 64,  576,  1);
        packW(PW(4 + sx),      wf_fea_1, 64, 128, 1152, 0);

        // ----- L3 (32x32) -----
        toNHWC(a3, cat3, 32, 32, 64, 128, 0);
        toNHWC(b3, cat3, 32, 32, 64, 128, 64);
        conv(cat3, wf_oc1_3, PB(28 + sx), nullptr, t3a, 32, 32, 64, 0, 1, 64, 0);
        conv(t3a, wf_oc2_3, PB(30 + sx), nullptr, off3, 32, 32, 64, 1, 1, 64, 0);
        conv(off3, wf_doff_3, PB(25 + 2 * sx), nullptr, o216_3, 32, 32, 216, 1, 0, 216, 0);
        gather(cat3, o216_3, 32, 32);
        conv(val, wf_dcn_3, PB(24 + 2 * sx), nullptr, fea3, 32, 32, 64, 2, 1, 64, 0);

        // ----- L2 (64x64) -----
        toNHWC(a2, cat2, 64, 64, 64, 128, 0);
        toNHWC(b2, cat2, 64, 64, 64, 128, 64);
        conv(cat2, wf_oc1_2, PB(18 + sx), nullptr, t2cat, 64, 64, 64, 0, 1, 128, 0);
        up2(off3, 32, 32, t2cat, 128, 64, 2.0f);
        conv(t2cat, wf_oc2_2, PB(20 + sx), nullptr, t2b, 64, 64, 64, 0, 1, 64, 0);
        conv(t2b, wf_oc3_2, PB(22 + sx), nullptr, off2, 64, 64, 64, 1, 1, 64, 0);
        conv(off2, wf_doff_2, PB(13 + 2 * sx), nullptr, o216_2, 64, 64, 216, 1, 0, 216, 0);
        gather(cat2, o216_2, 64, 64);
        conv(val, wf_dcn_2, PB(12 + 2 * sx), nullptr, dcat2, 64, 64, 64, 2, 0, 128, 0);
        up2(fea3, 32, 32, dcat2, 128, 64, 1.0f);
        conv(dcat2, wf_fea_2, PB(16 + sx), nullptr, fea2, 64, 64, 64, 0, 1, 64, 0);

        // ----- L1 (128x128) -----
        toNHWC(a1, cat1, 128, 128, 64, 128, 0);
        toNHWC(b1, cat1, 128, 128, 64, 128, 64);
        conv(cat1, wf_oc1_1, PB(6 + sx), nullptr, t1cat, 128, 128, 64, 0, 1, 128, 0);
        up2(off2, 64, 64, t1cat, 128, 64, 2.0f);
        conv(t1cat, wf_oc2_1, PB(8 + sx), nullptr, t1b, 128, 128, 64, 0, 1, 64, 0);
        conv(t1b, wf_oc3_1, PB(10 + sx), nullptr, off1, 128, 128, 64, 1, 1, 64, 0);
        conv(off1, wf_doff_1, PB(1 + 2 * sx), nullptr, o216_1, 128, 128, 216, 1, 0, 216, 0);
        gather(cat1, o216_1, 128, 128);
        conv(val, wf_dcn_1, PB(0 + 2 * sx), nullptr, dcat1, 128, 128, 64, 2, 0, 128, 0);
        up2(fea2, 64, 64, dcat1, 128, 64, 1.0f);
        conv(dcat1, wf_fea_1, PB(4 + sx), (float*)d_out, nullptr,
             128, 128, 64, 0, 0, 128, sx * 64);
    }
}
